// LSHAttention_48498770706949
// MI455X (gfx1250) — matrix operand order repaired and checked by reading, not compiled, not run on hardware
//
#include <hip/hip_runtime.h>
#include <math.h>

// ---------------- problem constants ----------------
#define S_LEN   8192
#define DIM     64
#define NHASH   8
#define NBUCK   128          // buckets per hash round (S/64)
#define NBTOT   1024         // NHASH * NBUCK
#define NKEY    65536        // NHASH * S_LEN  (keys per batch)
#define NCHUNK  1024         // chunks per batch
#define CHROWS  64           // rows per chunk
#define SEGSZ   256          // counting-sort segment size
#define NSEG    256          // NKEY / SEGSZ
#define NBATCH  16

typedef __attribute__((ext_vector_type(16))) _Float16 v16h;
typedef __attribute__((ext_vector_type(8)))  float    v8f;

// =====================================================================
// Kernel 1: LSH hashing with WMMA.
// Per (b,h): rotated = qk(8192x64) @ rot(64x64); bucket = argmax[rot,-rot].
// Block = 256 threads (8 waves) = 128 tokens; rotations staged transposed
// in LDS so WMMA B-fragments are contiguous 16-half runs.
// =====================================================================
__global__ __launch_bounds__(256) void hash_kernel(const float* __restrict__ qk,
                                                   const float* __restrict__ rot,
                                                   int* __restrict__ bkt) {
    __shared__ _Float16 sRotT[64 * 64];   // [i][f]  (transposed rotations)
    __shared__ _Float16 sQ[128 * 64];     // [tok][f]
    __shared__ float    sRt[128 * 64];    // rotated [tok][i] (f32)

    const int b = blockIdx.z, h = blockIdx.y;
    const int t0 = blockIdx.x * 128;
    const int tid = threadIdx.x, lane = tid & 31, wv = tid >> 5;
    const int hi = (lane >> 4) & 1, l16 = lane & 15;

    // stage rotations transposed: sRotT[i*64+f] = rot[b,f,h,i]
    for (int l = tid; l < 4096; l += 256) {
        int f = l >> 6, i = l & 63;
        sRotT[i * 64 + f] = (_Float16)rot[((b * 64 + f) * NHASH + h) * 64 + i];
    }
    // stage q tile: thread -> row tid/2, half (tid&1)*32
    {
        int row = tid >> 1, hf = (tid & 1) * 32;
        const float4* src = (const float4*)(qk + ((b * S_LEN + t0 + row) << 6) + hf);
#pragma unroll
        for (int cc = 0; cc < 8; ++cc) {
            float4 x = src[cc];
            sQ[row * 64 + hf + cc * 4 + 0] = (_Float16)x.x;
            sQ[row * 64 + hf + cc * 4 + 1] = (_Float16)x.y;
            sQ[row * 64 + hf + cc * 4 + 2] = (_Float16)x.z;
            sQ[row * 64 + hf + cc * 4 + 3] = (_Float16)x.w;
        }
    }
    __syncthreads();

    // wave wv computes rotated rows [wv*16, wv*16+16) x all 64 columns
    const int rowB = wv * 16;
    v16h a0, a1;
#pragma unroll
    for (int e = 0; e < 16; ++e) {
        int k0 = e + 16 * hi;                    // A 16x32 f16 layout
        a0[e] = sQ[(rowB + l16) * 64 + k0];
        a1[e] = sQ[(rowB + l16) * 64 + 32 + k0];
    }
    for (int nt = 0; nt < 4; ++nt) {
        v8f acc = {};
        v16h b0, b1;
#pragma unroll
        for (int e = 0; e < 16; ++e) {
            int kk = e + 16 * hi;                // B 32x16 f16 layout
            b0[e] = sRotT[(nt * 16 + l16) * 64 + kk];
            b1[e] = sRotT[(nt * 16 + l16) * 64 + 32 + kk];
        }
        acc = __builtin_amdgcn_wmma_f32_16x16x32_f16(false, a0, false, b0,
                                                     (short)0, acc, false, false);
        acc = __builtin_amdgcn_wmma_f32_16x16x32_f16(false, a1, false, b1,
                                                     (short)0, acc, false, false);
#pragma unroll
        for (int r = 0; r < 8; ++r)
            sRt[(rowB + r + 8 * hi) * 64 + nt * 16 + l16] = acc[r];
    }
    __syncthreads();

    // argmax over [rot, -rot] per token (first-occurrence tie-break)
    if (tid < 128) {
        float mx = -3.4e38f, mn = 3.4e38f;
        int imx = 0, imn = 0;
        for (int i = 0; i < 64; ++i) {
            float d = sRt[tid * 64 + i];
            if (d > mx) { mx = d; imx = i; }     // strict > : first max
            if (d < mn) { mn = d; imn = i; }     // strict < : first min
        }
        int bucket = (mx >= -mn) ? imx : (64 + imn);
        bkt[b * NKEY + h * S_LEN + (t0 + tid)] = h * NBUCK + bucket;
    }
}

// =====================================================================
// Kernel 2: per-(batch,segment) histogram of 1024 buckets.
// counts layout per batch: [bucket][segment]  (bucket-major => stable)
// =====================================================================
__global__ __launch_bounds__(256) void hist_kernel(const int* __restrict__ bkt,
                                                   int* __restrict__ counts) {
    __shared__ int hist[NBTOT];
    const int seg = blockIdx.x, b = blockIdx.y, tid = threadIdx.x;
    for (int i = tid; i < NBTOT; i += 256) hist[i] = 0;
    __syncthreads();
    int key = bkt[b * NKEY + seg * SEGSZ + tid];
    atomicAdd(&hist[key], 1);
    __syncthreads();
    for (int i = tid; i < NBTOT; i += 256)
        counts[b * (NBTOT * NSEG) + i * NSEG + seg] = hist[i];
}

// =====================================================================
// Kernel 3: per-batch exclusive prefix sum over the 262144 counts
// (bucket-major, segment-minor). One 1024-thread block per batch.
// =====================================================================
__global__ __launch_bounds__(1024) void scan_kernel(int* __restrict__ counts) {
    __shared__ int part[1024];
    const int b = blockIdx.x, tid = threadIdx.x;
    int* base = counts + b * (NBTOT * NSEG);
    const int off0 = tid * 256;
    int s = 0;
    for (int i = 0; i < 256; ++i) s += base[off0 + i];
    part[tid] = s;
    __syncthreads();
    for (int off = 1; off < 1024; off <<= 1) {
        int v = part[tid];
        if (tid >= off) v += part[tid - off];
        __syncthreads();
        part[tid] = v;
        __syncthreads();
    }
    int run = (tid == 0) ? 0 : part[tid - 1];
    for (int i = 0; i < 256; ++i) {
        int c = base[off0 + i];
        base[off0 + i] = run;
        run += c;
    }
}

// =====================================================================
// Kernel 4: stable scatter.  sticker[b][dest] = j  (j = h*S + t)
// =====================================================================
__global__ __launch_bounds__(256) void scatter_kernel(const int* __restrict__ bkt,
                                                      const int* __restrict__ counts,
                                                      int* __restrict__ sticker) {
    __shared__ int lb[SEGSZ];
    const int seg = blockIdx.x, b = blockIdx.y, tid = threadIdx.x;
    const int j = seg * SEGSZ + tid;
    const int mybkt = bkt[b * NKEY + j];
    lb[tid] = mybkt;
    __syncthreads();
    int rank = 0;
    for (int u = 0; u < tid; ++u) rank += (lb[u] == mybkt) ? 1 : 0;
    int dest = counts[b * (NBTOT * NSEG) + mybkt * NSEG + seg] + rank;
    sticker[b * NKEY + dest] = j;
}

// =====================================================================
// Kernel 5: chunked attention with f16 WMMA.
// PASS 0: write per-row logsumexp to logits[b][h][t].
// PASS 1: fold softmax-over-hash weight into probs, atomicAdd output.
// One workgroup (4 waves / 128 threads) per (batch, chunk).
// V is staged transposed so PV B-fragments are contiguous.
// =====================================================================
struct QKBuf { _Float16 q[64 * 64]; _Float16 k[128 * 64]; };
union UQKP { QKBuf s; _Float16 p[64 * 128]; };   // probs alias dead q/k

template <int PASS>
__global__ __launch_bounds__(128) void attn_kernel(const float* __restrict__ qk,
                                                   const float* __restrict__ vg,
                                                   const int* __restrict__ sticker,
                                                   const float* __restrict__ wgt,
                                                   float* __restrict__ logits,
                                                   float* __restrict__ out) {
    __shared__ UQKP u;
    __shared__ _Float16 sDots[64 * 128];
    __shared__ _Float16 sVt[64 * 128];   // [d][key]  (transposed V)
    __shared__ int sJ[128];

    const int b = blockIdx.y, c = blockIdx.x, tid = threadIdx.x;
    const int lane = tid & 31, wv = tid >> 5;
    const int hi = (lane >> 4) & 1, l16 = lane & 15;

    // sorted tickers: columns [0,64) = own chunk, [64,128) = previous (wrap)
    if (tid < 64) {
        sJ[tid] = sticker[b * NKEY + c * CHROWS + tid];
    } else {
        int cp = (c + NCHUNK - 1) & (NCHUNK - 1);
        sJ[tid] = sticker[b * NKEY + cp * CHROWS + (tid - 64)];
    }
    __syncthreads();

    // gather + stage: each thread owns one k row; threads<64 also own q row
    {
        int j = sJ[tid];
        int t = j & (S_LEN - 1);
        const float4* row = (const float4*)(qk + ((b * S_LEN + t) << 6));
        float buf[64];
        float ss = 0.f;
#pragma unroll
        for (int cc = 0; cc < 16; ++cc) {
            float4 x = row[cc];
            buf[cc * 4 + 0] = x.x; buf[cc * 4 + 1] = x.y;
            buf[cc * 4 + 2] = x.z; buf[cc * 4 + 3] = x.w;
            ss += x.x * x.x + x.y * x.y + x.z * x.z + x.w * x.w;
        }
        float sc = 1.f / fmaxf(sqrtf(ss), 1e-12f);
#pragma unroll
        for (int f = 0; f < 64; ++f) u.s.k[tid * 64 + f] = (_Float16)(buf[f] * sc);
        if (PASS == 1) {   // stage V transposed: sVt[d*128 + key]
            const float4* vrow = (const float4*)(vg + ((b * S_LEN + t) << 6));
#pragma unroll
            for (int cc = 0; cc < 16; ++cc) {
                float4 x = vrow[cc];
                sVt[(cc * 4 + 0) * 128 + tid] = (_Float16)x.x;
                sVt[(cc * 4 + 1) * 128 + tid] = (_Float16)x.y;
                sVt[(cc * 4 + 2) * 128 + tid] = (_Float16)x.z;
                sVt[(cc * 4 + 3) * 128 + tid] = (_Float16)x.w;
            }
        }
        if (tid < 64) {  // q = raw row * D^-0.5 (scale folded into A operand)
#pragma unroll
            for (int f = 0; f < 64; ++f) u.s.q[tid * 64 + f] = (_Float16)(buf[f] * 0.125f);
        }
    }
    __syncthreads();

    // ---- dots = q @ k^T : wave wv owns 16 q rows, 8 column tiles, K=64 ----
    const int rowB = wv * 16;
    v16h a0, a1;
#pragma unroll
    for (int e = 0; e < 16; ++e) {
        int k0 = e + 16 * hi;                    // A 16x32 f16 layout
        a0[e] = u.s.q[(rowB + l16) * 64 + k0];
        a1[e] = u.s.q[(rowB + l16) * 64 + 32 + k0];
    }
    for (int jt = 0; jt < 8; ++jt) {
        v8f acc = {};
        v16h b0, b1;
#pragma unroll
        for (int e = 0; e < 16; ++e) {
            int kk = e + 16 * hi;                // B 32x16 f16 layout
            b0[e] = u.s.k[(jt * 16 + l16) * 64 + kk];
            b1[e] = u.s.k[(jt * 16 + l16) * 64 + 32 + kk];
        }
        acc = __builtin_amdgcn_wmma_f32_16x16x32_f16(false, a0, false, b0,
                                                     (short)0, acc, false, false);
        acc = __builtin_amdgcn_wmma_f32_16x16x32_f16(false, a1, false, b1,
                                                     (short)0, acc, false, false);
#pragma unroll
        for (int r = 0; r < 8; ++r) {            // C/D layout: M = r+8*hi, N = l16
            int row = rowB + r + 8 * hi;
            int col = jt * 16 + l16;
            float val = acc[r];
            if ((sJ[row] & (S_LEN - 1)) == (sJ[col] & (S_LEN - 1))) val = -5e4f;
            sDots[row * 128 + col] = (_Float16)val;
        }
    }
    __syncthreads();

    // ---- per-row softmax / logsumexp (threads 0..63, one row each) ----
    if (tid < 64) {
        const int row = tid;
        float m = -3.4e38f;
        for (int jj = 0; jj < 128; ++jj)
            m = fmaxf(m, (float)sDots[row * 128 + jj]);
        float sum = 0.f;
        for (int jj = 0; jj < 128; ++jj)
            sum += expf((float)sDots[row * 128 + jj] - m);
        float lse = m + logf(sum);
        int j0 = sJ[row];
        int h = j0 >> 13;                 // j0 / S_LEN
        int t = j0 & (S_LEN - 1);
        if (PASS == 0) {
            logits[(b * NHASH + h) * S_LEN + t] = lse;
        } else {
            float wt = wgt[(b * NHASH + h) * S_LEN + t];
            for (int jj = 0; jj < 128; ++jj)
                u.p[row * 128 + jj] =
                    (_Float16)(expf((float)sDots[row * 128 + jj] - lse) * wt);
        }
    }

    if (PASS == 1) {
        __syncthreads();
        // ---- out += probs(64x128) @ v(128x64), K in 4 chunks of 32 ----
        v16h pa[4];
#pragma unroll
        for (int kt = 0; kt < 4; ++kt) {
#pragma unroll
            for (int e = 0; e < 16; ++e) {
                int k0 = e + 16 * hi;
                pa[kt][e] = u.p[(rowB + l16) * 128 + kt * 32 + k0];
            }
        }
        for (int nt = 0; nt < 4; ++nt) {
            v8f acc = {};
            for (int kt = 0; kt < 4; ++kt) {
                v16h bf;
#pragma unroll
                for (int e = 0; e < 16; ++e)   // contiguous in sVt
                    bf[e] = sVt[(nt * 16 + l16) * 128 + kt * 32 + e + 16 * hi];
                acc = __builtin_amdgcn_wmma_f32_16x16x32_f16(false, pa[kt], false, bf,
                                                             (short)0, acc, false, false);
            }
#pragma unroll
            for (int r = 0; r < 8; ++r) {
                int row = rowB + r + 8 * hi;
                int col = nt * 16 + l16;
                int t = sJ[row] & (S_LEN - 1);
                atomicAdd(&out[((b * S_LEN + t) << 6) + col], acc[r]);
            }
        }
    }
}

// =====================================================================
// Kernel 6: softmax over the 8 hash rounds -> per-(b,h,t) weight.
// =====================================================================
__global__ __launch_bounds__(256) void weights_kernel(const float* __restrict__ logits,
                                                      float* __restrict__ wgt) {
    int idx = blockIdx.x * 256 + threadIdx.x;   // = b*S + t
    int b = idx >> 13;
    int t = idx & (S_LEN - 1);
    float l[NHASH];
    float m = -3.4e38f;
#pragma unroll
    for (int h = 0; h < NHASH; ++h) {
        l[h] = logits[(b * NHASH + h) * S_LEN + t];
        m = fmaxf(m, l[h]);
    }
    float sum = 0.f;
#pragma unroll
    for (int h = 0; h < NHASH; ++h) sum += expf(l[h] - m);
    float lse = m + logf(sum);
#pragma unroll
    for (int h = 0; h < NHASH; ++h)
        wgt[(b * NHASH + h) * S_LEN + t] = expf(l[h] - lse);
}

__global__ __launch_bounds__(256) void zero_kernel(float* __restrict__ p, int n) {
    int i = blockIdx.x * 256 + threadIdx.x;
    if (i < n) p[i] = 0.f;
}

// =====================================================================
// Launch: hash -> hist -> scan -> scatter -> attn(logits) -> weights
//         -> zero(out) -> attn(output, atomic accumulate)
// Workspace (32 MiB): bkt 4M | counts 16M | sticker 4M | logits 4M | wgt 4M
// =====================================================================
extern "C" void kernel_launch(void* const* d_in, const int* in_sizes, int n_in,
                              void* d_out, int out_size, void* d_ws, size_t ws_size,
                              hipStream_t stream) {
    (void)in_sizes; (void)n_in; (void)out_size; (void)ws_size;
    const float* qk  = (const float*)d_in[0];
    const float* v   = (const float*)d_in[1];
    const float* rot = (const float*)d_in[2];
    float* out = (float*)d_out;

    char* ws = (char*)d_ws;
    int*   bkt     = (int*)(ws);
    int*   counts  = (int*)(ws + (size_t)4  * 1024 * 1024);
    int*   sticker = (int*)(ws + (size_t)20 * 1024 * 1024);
    float* logits  = (float*)(ws + (size_t)24 * 1024 * 1024);
    float* wgt     = (float*)(ws + (size_t)28 * 1024 * 1024);

    hash_kernel<<<dim3(S_LEN / 128, NHASH, NBATCH), 256, 0, stream>>>(qk, rot, bkt);
    hist_kernel<<<dim3(NSEG, NBATCH), 256, 0, stream>>>(bkt, counts);
    scan_kernel<<<dim3(NBATCH), 1024, 0, stream>>>(counts);
    scatter_kernel<<<dim3(NSEG, NBATCH), 256, 0, stream>>>(bkt, counts, sticker);

    attn_kernel<0><<<dim3(NCHUNK, NBATCH), 128, 0, stream>>>(
        qk, v, sticker, nullptr, logits, nullptr);

    weights_kernel<<<dim3(NBATCH * S_LEN / 256), 256, 0, stream>>>(logits, wgt);

    int nout = NBATCH * S_LEN * DIM;
    zero_kernel<<<dim3((nout + 255) / 256), 256, 0, stream>>>(out, nout);

    attn_kernel<1><<<dim3(NCHUNK, NBATCH), 128, 0, stream>>>(
        qk, v, sticker, wgt, logits, out);
}
